// ConditionalRandomField_11570641896014
// MI455X (gfx1250) — compile-verified, matrix operand order checked by reading, not run on hardware
//
#include <hip/hip_runtime.h>

typedef __attribute__((ext_vector_type(16))) _Float16 v16h;
typedef __attribute__((ext_vector_type(8)))  float    v8f;

#define K_TAGS 128
#define DDIM   512
#define BATCH  256
#define TIN    126
#define NPOS   (BATCH * TIN)
#define NEGF   (-1e30f)

#define EF_STRIDE   520   // f32 staging row stride (2080 B, 16B-aligned, bank-spread)
#define ELDS_STRIDE 528   // f16 row stride (1056 B, 16B-aligned, bank-spread)
#define ATS         144   // A^T LDS row stride in f16 (288 B, 16B-aligned)

// low 32 bits of a flat pointer into LDS = LDS byte offset (aperture layout)
__device__ __forceinline__ unsigned lds_off(const void* p) {
    return (unsigned)(uintptr_t)p;
}

// CDNA5 async global->LDS copy (16 B per lane), tracked by ASYNCcnt
__device__ __forceinline__ void async_g2l_b128(unsigned lds_addr, const void* gaddr) {
    asm volatile("global_load_async_to_lds_b128 %0, %1, off"
                 :: "v"(lds_addr), "v"(gaddr) : "memory");
}
__device__ __forceinline__ void wait_async0() {
    asm volatile("s_wait_asynccnt 0" ::: "memory");
}

__device__ __forceinline__ v16h load_frag(const _Float16* p0, const _Float16* p1) {
    v16h a;
    ((uint4*)&a)[0] = *(const uint4*)p0;   // K = kb .. kb+7
    ((uint4*)&a)[1] = *(const uint4*)p1;   // K = kb+16 .. kb+23
    return a;
}

// ---------------- Kernel A: precompute f16 operands -------------------------
// ThetaB_h[t*512+d] = f16(ThetaB)          (acts as B^T for the emission GEMM)
// At_h[t*128+p]     = f16(exp(WA[p,t]))    with BOS column (t==126) zeroed
__global__ __launch_bounds__(256) void prep_kernel(
    const float* __restrict__ ThetaB, const float* __restrict__ WA,
    _Float16* __restrict__ ThetaB_h, _Float16* __restrict__ At_h) {
    int tid = blockIdx.x * blockDim.x + threadIdx.x;       // 65536 threads
    ThetaB_h[tid] = (_Float16)ThetaB[tid];
    if (tid < K_TAGS * K_TAGS) {
        int t = tid >> 7, p = tid & 127;
        float v = (t == K_TAGS - 2) ? 0.0f : __expf(WA[p * K_TAGS + t]);
        At_h[tid] = (_Float16)v;
    }
}

// ---------------- Kernel B: gathered emission GEMM --------------------------
// emits[pos, t] = dot(E[words[pos]], ThetaB[t]);  cols t in {126,127} -> NEG
__global__ __launch_bounds__(256) void emit_kernel(
    const float* __restrict__ E, const int* __restrict__ words,
    const _Float16* __restrict__ ThetaB_h, float* __restrict__ emits) {
    __shared__ __align__(16) float    Ef[16 * EF_STRIDE];     // f32 staging
    __shared__ __align__(16) _Float16 Elds[16 * ELDS_STRIDE]; // f16 A-operand
    const int tid = threadIdx.x;
    const int posBase = blockIdx.x * 16;

    { // async-gather 16 E rows (f32) straight into LDS, then convert own bytes
        int r   = tid >> 4;
        int c16 = tid & 15;                       // thread owns 128 B of row r
        int w   = words[posBase + r];
        const char* gsrc = (const char*)(E + (size_t)w * DDIM) + c16 * 128;
        unsigned dst = lds_off(Ef) + r * (EF_STRIDE * 4) + c16 * 128;
#pragma unroll
        for (int i = 0; i < 8; ++i)
            async_g2l_b128(dst + i * 16, gsrc + i * 16);
        wait_async0();                            // own wave's copies landed
        const float4* s4 = (const float4*)(Ef + r * EF_STRIDE + c16 * 32);
        _Float16* d = Elds + r * ELDS_STRIDE + c16 * 32;
#pragma unroll
        for (int i = 0; i < 8; ++i) {
            float4 f = s4[i];
            d[i * 4 + 0] = (_Float16)f.x; d[i * 4 + 1] = (_Float16)f.y;
            d[i * 4 + 2] = (_Float16)f.z; d[i * 4 + 3] = (_Float16)f.w;
        }
    }
    __syncthreads();

    const int wave = tid >> 5, lane = tid & 31;
    const int ln = lane & 15, h = lane >> 4;
    const int n0 = wave * 16;                         // 8 waves -> 8 N-tiles
    v8f acc = {};
    const _Float16* Arow = Elds + ln * ELDS_STRIDE;             // position row
    const _Float16* Brow = ThetaB_h + (size_t)(n0 + ln) * DDIM; // tag row (B^T)
#pragma unroll
    for (int kc = 0; kc < DDIM / 32; ++kc) {
        int kb = kc * 32 + h * 8;
        v16h a = load_frag(Arow + kb, Arow + kb + 16);
        v16h b = load_frag(Brow + kb, Brow + kb + 16);
        acc = __builtin_amdgcn_wmma_f32_16x16x32_f16(false, a, false, b,
                                                     (short)0, acc, false, false);
    }
#pragma unroll
    for (int r = 0; r < 8; ++r) {                    // D: lane = col, vgpr = row
        int m = h * 8 + r, col = n0 + ln;
        float v = (col >= K_TAGS - 2) ? NEGF : acc[r];
        emits[(size_t)(posBase + m) * K_TAGS + col] = v;
    }
}

// ---------------- Kernel C: rescaled forward recursion ----------------------
// Each WG owns 16 batch rows; A^T, alpha, P, S live in LDS for all 126 steps.
__global__ __launch_bounds__(256) void forward_kernel(
    const float* __restrict__ WA, const float* __restrict__ emits,
    const _Float16* __restrict__ At_h, float* __restrict__ logZ) {
    __shared__ __align__(16) _Float16 Ats[K_TAGS * ATS];
    __shared__ __align__(16) _Float16 P[16 * ATS];
    __shared__ float la[16][K_TAGS];
    __shared__ float S[16][K_TAGS];
    __shared__ float red[16][16];

    const int tid = threadIdx.x;
    const int b0  = blockIdx.x * 16;
    const int r   = tid >> 4;        // batch row within WG
    const int l16 = tid & 15;

    { // async copy A^T (f16, 128x128, global stride 256 B) into padded LDS
        unsigned atbase = lds_off(Ats);
#pragma unroll
        for (int k = 0; k < 8; ++k) {
            int c = tid + k * 256;               // 2048 16-byte chunks
            int row = c >> 4, part = c & 15;
            async_g2l_b128(atbase + row * (ATS * 2) + part * 16,
                           (const char*)At_h + row * 256 + part * 16);
        }
        wait_async0();
    }
#pragma unroll
    for (int i = 0; i < 8; ++i) {                            // alpha0 = onehot(bos)
        int c = l16 * 8 + i;
        la[r][c] = (c == K_TAGS - 2) ? 0.0f : NEGF;
    }
    __syncthreads();

    const int wave = tid >> 5, lane = tid & 31;
    const int ln = lane & 15, h = lane >> 4, n0 = wave * 16;

    for (int j = 0; j < TIN; ++j) {
        // 1) per-row max
        float pm = NEGF;
#pragma unroll
        for (int i = 0; i < 8; ++i) pm = fmaxf(pm, la[r][l16 * 8 + i]);
        red[r][l16] = pm;
        __syncthreads();
        float m = NEGF;
#pragma unroll
        for (int i = 0; i < 16; ++i) m = fmaxf(m, red[r][i]);
        // 2) P = exp(alpha - m) -> f16
#pragma unroll
        for (int i = 0; i < 8; ++i) {
            int c = l16 * 8 + i;
            P[r * ATS + c] = (_Float16)__expf(la[r][c] - m);
        }
        __syncthreads();
        // 3) S = P @ A  (each wave: one 16x16 tile, K=128 -> 4 WMMA)
        v8f acc = {};
        const _Float16* Prow = P + ln * ATS;
        const _Float16* Brow = Ats + (n0 + ln) * ATS;
#pragma unroll
        for (int kc = 0; kc < K_TAGS / 32; ++kc) {
            int kb = kc * 32 + h * 8;
            v16h a = load_frag(Prow + kb, Prow + kb + 16);
            v16h b = load_frag(Brow + kb, Brow + kb + 16);
            acc = __builtin_amdgcn_wmma_f32_16x16x32_f16(false, a, false, b,
                                                         (short)0, acc, false, false);
        }
#pragma unroll
        for (int rr = 0; rr < 8; ++rr) S[h * 8 + rr][n0 + ln] = acc[rr];
        __syncthreads();
        // 4) alpha' = m + log(S) + emit   (S==0 on masked BOS col -> NEG)
#pragma unroll
        for (int i = 0; i < 8; ++i) {
            int c = l16 * 8 + i;
            float s = S[r][c];
            float e = emits[((size_t)(b0 + r) * TIN + j) * K_TAGS + c];
            la[r][c] = (s > 0.0f) ? (m + __logf(s) + e) : NEGF;
        }
        __syncthreads();
    }

    // final: logZ = logsumexp_p(alpha[p] + WA[p, eos])
    float v[8], pm = NEGF;
#pragma unroll
    for (int i = 0; i < 8; ++i) {
        int p = l16 * 8 + i;
        v[i] = la[r][p] + WA[p * K_TAGS + (K_TAGS - 1)];
        pm = fmaxf(pm, v[i]);
    }
    red[r][l16] = pm;
    __syncthreads();
    float m2 = NEGF;
#pragma unroll
    for (int i = 0; i < 16; ++i) m2 = fmaxf(m2, red[r][i]);
    float sum = 0.0f;
#pragma unroll
    for (int i = 0; i < 8; ++i) sum += __expf(v[i] - m2);
    __syncthreads();
    red[r][l16] = sum;
    __syncthreads();
    if (l16 == 0) {
        float tot = 0.0f;
#pragma unroll
        for (int i = 0; i < 16; ++i) tot += red[r][i];
        logZ[b0 + r] = m2 + __logf(tot);
    }
}

extern "C" void kernel_launch(void* const* d_in, const int* in_sizes, int n_in,
                              void* d_out, int out_size, void* d_ws, size_t ws_size,
                              hipStream_t stream) {
    const float* ThetaB = (const float*)d_in[0];   // (128, 512)
    const float* WA     = (const float*)d_in[1];   // (128, 128)
    const float* E      = (const float*)d_in[2];   // (50000, 512)
    const int*   words  = (const int*)d_in[3];     // (256, 126)
    float* logZ = (float*)d_out;                   // (256,)

    char* ws = (char*)d_ws;
    _Float16* ThetaB_h = (_Float16*)ws;                       // 128 KB
    _Float16* At_h     = (_Float16*)(ws + 131072);            // 32 KB
    float*    emits    = (float*)(ws + 131072 + 32768);       // 16.5 MB

    prep_kernel<<<256, 256, 0, stream>>>(ThetaB, WA, ThetaB_h, At_h);
    emit_kernel<<<NPOS / 16, 256, 0, stream>>>(E, words, ThetaB_h, emits);
    forward_kernel<<<BATCH / 16, 256, 0, stream>>>(WA, emits, At_h, logZ);
}
